// GrapnTransformer_64209761075689
// MI455X (gfx1250) — matrix operand order repaired and checked by reading, not compiled, not run on hardware
//
#include <hip/hip_runtime.h>

#define HID   64
#define HEADS 4
#define DH    16

typedef __attribute__((ext_vector_type(16))) _Float16 v16h;
typedef __attribute__((ext_vector_type(8)))  _Float16 v8h;
typedef __attribute__((ext_vector_type(8)))  float    v8f;
typedef __attribute__((ext_vector_type(4)))  int      i4v;

// ---------------------------------------------------------------------------
// gfx1250 async LDS staging (guarded: falls back to plain vector copy).
// ROCm clang-22 declares the builtin with generic `int4*` pointer params
// (probe-confirmed via diagnostic), not addrspace-qualified void*.
// ---------------------------------------------------------------------------
#if defined(__HIP_DEVICE_COMPILE__) && __has_builtin(__builtin_amdgcn_global_load_async_to_lds_b128)
#define CDNA5_ASYNC_LDS 1
#endif

__device__ __forceinline__ void stage16(const void* g, void* l) {
#ifdef CDNA5_ASYNC_LDS
  __builtin_amdgcn_global_load_async_to_lds_b128((i4v*)g, (i4v*)l, 0, 0);
#else
  *(float4*)l = *(const float4*)g;
#endif
}
__device__ __forceinline__ void wait_async() {
#ifdef CDNA5_ASYNC_LDS
  asm volatile("s_wait_asynccnt 0x0" ::: "memory");
#endif
}

// ---------------------------------------------------------------------------
// Weight conversion: W[64][fin] f32 -> WhT[kpad][64] f16 (k-major, padded)
// ---------------------------------------------------------------------------
__global__ void k_cvt_weight(const float* __restrict__ W, _Float16* __restrict__ whT,
                             int fin, int kpad) {
  int idx = blockIdx.x * blockDim.x + threadIdx.x;
  if (idx >= kpad * HID) return;
  int k = idx / HID, o = idx - k * HID;
  whT[idx] = (_Float16)((k < fin) ? W[o * fin + k] : 0.0f);
}

// x[N][15] f32 -> xh[N][32] f16 zero-padded (contiguous f16 tiles for async DMA)
__global__ void k_cvt_x(const float* __restrict__ x, _Float16* __restrict__ xh, int N) {
  int idx = blockIdx.x * blockDim.x + threadIdx.x;
  if (idx >= N * 32) return;
  int n = idx >> 5, k = idx & 31;
  xh[idx] = (_Float16)((k < 15) ? x[(size_t)n * 15 + k] : 0.0f);
}

// relu(h) f32 -> f16 copy (layer-2 GEMM input)
__global__ void k_relu_cvt(const float* __restrict__ a, _Float16* __restrict__ xh, int n) {
  int idx = blockIdx.x * blockDim.x + threadIdx.x;
  if (idx < n) xh[idx] = (_Float16)fmaxf(a[idx], 0.0f);
}

__global__ void k_fill(float* __restrict__ p, float v, int n) {
  int idx = blockIdx.x * blockDim.x + threadIdx.x;
  if (idx < n) p[idx] = v;
}

// ---------------------------------------------------------------------------
// Persistent fused 4-way linear: out_w = in @ W_w^T + b_w, w in {Q,K,V,Skip}.
// Block = 128 thr = 4 waves; wave w owns weight matrix w, B fragments hoisted
// into registers once. Grid-stride over 16-row tiles; each tile's f16 rows are
// contiguous, staged to LDS via async-DMA, consumed by all 4 waves.
//
// Fragment layouts per CDNA5 ISA 7.12.2 (wave32):
//   A (16x32 f16): lane[0:15]->M; elem pairs->K with half-wave +8 split.
//   B (32x16 f16): lane&15->N; elems->K, same half-wave +8 split as A.
//   C/D (16x16 f32): vgpr r -> M = r + 8*(lane>=16), lane&15 -> N.
// ---------------------------------------------------------------------------
template<int KPAD>
__global__ __launch_bounds__(128) void k_linear4_wmma(
    const _Float16* __restrict__ xh,
    const _Float16* __restrict__ w0, const _Float16* __restrict__ w1,
    const _Float16* __restrict__ w2, const _Float16* __restrict__ w3,
    const float* __restrict__ b0, const float* __restrict__ b1,
    const float* __restrict__ b2, const float* __restrict__ b3,
    float* __restrict__ o0, float* __restrict__ o1,
    float* __restrict__ o2, float* __restrict__ o3,
    int nrows, int ntiles)
{
  __shared__ __align__(16) _Float16 xs[16 * KPAD];
  const int tid  = threadIdx.x;
  const int wave = tid >> 5;
  const int lane = tid & 31;
  const int half = lane >> 4;
  const int l15  = lane & 15;

  const _Float16* wht = (wave == 0) ? w0 : (wave == 1) ? w1 : (wave == 2) ? w2 : w3;
  const float*   bias = (wave == 0) ? b0 : (wave == 1) ? b1 : (wave == 2) ? b2 : b3;
  float*          out = (wave == 0) ? o0 : (wave == 1) ? o1 : (wave == 2) ? o2 : o3;

  // Hoist B fragments (weights live in registers for the whole kernel).
  v16h bf[KPAD / 32][4];
#pragma unroll
  for (int kc = 0; kc < KPAD / 32; kc++) {
#pragma unroll
    for (int ct = 0; ct < 4; ct++) {
      const _Float16* wp = wht + (size_t)(kc * 32 + half * 8) * HID + ct * 16 + l15;
#pragma unroll
      for (int s = 0; s < 8; s++) {
        bf[kc][ct][s]     = wp[(size_t)s * HID];
        bf[kc][ct][8 + s] = wp[(size_t)(16 + s) * HID];
      }
    }
  }
  const float bv0 = bias[0 * 16 + l15];
  const float bv1 = bias[1 * 16 + l15];
  const float bv2 = bias[2 * 16 + l15];
  const float bv3 = bias[3 * 16 + l15];

  constexpr int TILE_BYTES = 16 * KPAD * 2;   // one tile = 16 contiguous rows
  constexpr int NXFER      = TILE_BYTES / 16; // 16B transfers per tile (64/128)

  for (int tile = blockIdx.x; tile < ntiles; tile += gridDim.x) {
    const int row0 = tile * 16;
    const char* srcb = (const char*)xh + (size_t)row0 * (KPAD * 2);

    if (tid < NXFER) {
      stage16(srcb + tid * 16, (char*)xs + tid * 16);
      if (tile + (int)gridDim.x < ntiles)
        __builtin_prefetch(srcb + (size_t)gridDim.x * TILE_BYTES + tid * 16, 0, 1);
    }
    wait_async();
    __syncthreads();

    v8f acc[4] = {};
#pragma unroll
    for (int kc = 0; kc < KPAD / 32; kc++) {
      const int kb = kc * 32;
      v16h a;
#pragma unroll
      for (int v = 0; v < 4; v++) {
        int k0 = kb + half * 8 + 2 * v;       // elems 0..7: 8 contiguous halves
        a[2 * v]         = xs[l15 * KPAD + k0];
        a[2 * v + 1]     = xs[l15 * KPAD + k0 + 1];
        int k1 = k0 + 16;                     // elems 8..15: next 8 contiguous
        a[8 + 2 * v]     = xs[l15 * KPAD + k1];
        a[8 + 2 * v + 1] = xs[l15 * KPAD + k1 + 1];
      }
#pragma unroll
      for (int ct = 0; ct < 4; ct++) {
        acc[ct] = __builtin_amdgcn_wmma_f32_16x16x32_f16(
                      false, a, false, bf[kc][ct], (short)0, acc[ct], false, false);
      }
    }

    if (row0 + 16 <= nrows) {           // uniform fast path: no exec juggling
#pragma unroll
      for (int r = 0; r < 8; r++) {
        float* op = out + (size_t)(row0 + r + 8 * half) * HID + l15;
        op[0]  = acc[0][r] + bv0;
        op[16] = acc[1][r] + bv1;
        op[32] = acc[2][r] + bv2;
        op[48] = acc[3][r] + bv3;
      }
    } else {                             // ragged last tile only
#pragma unroll
      for (int r = 0; r < 8; r++) {
        int row = row0 + r + 8 * half;
        if (row < nrows) {
          float* op = out + (size_t)row * HID + l15;
          op[0]  = acc[0][r] + bv0;
          op[16] = acc[1][r] + bv1;
          op[32] = acc[2][r] + bv2;
          op[48] = acc[3][r] + bv3;
        }
      }
    }
    __syncthreads();
  }
}

// ---------------------------------------------------------------------------
// Edge attention phase (segment softmax over dst via atomics).
// ---------------------------------------------------------------------------
__device__ inline void atomic_max_f32(float* addr, float val) {
  if (val >= 0.0f) atomicMax((int*)addr, __float_as_int(val));
  else             atomicMin((unsigned int*)addr, __float_as_uint(val));
}

__global__ void k_edge_score(const float* __restrict__ Q, const float* __restrict__ K,
                             const int* __restrict__ src, const int* __restrict__ dst,
                             float* __restrict__ sc, float* __restrict__ m, int E)
{
  int idx = blockIdx.x * blockDim.x + threadIdx.x;
  if (idx >= E * HEADS) return;
  int e = idx >> 2, h = idx & 3;
  int s = src[e], d = dst[e];
  const float4* qp = (const float4*)(Q + (size_t)d * HID + h * DH);
  const float4* kp = (const float4*)(K + (size_t)s * HID + h * DH);
  float acc = 0.0f;
#pragma unroll
  for (int j = 0; j < 4; j++) {
    float4 qv = qp[j], kv = kp[j];
    acc += qv.x * kv.x + qv.y * kv.y + qv.z * kv.z + qv.w * kv.w;
  }
  acc *= 0.25f;  // 1/sqrt(DH)
  sc[idx] = acc;
  atomic_max_f32(&m[(size_t)d * HEADS + h], acc);
}

__global__ void k_edge_exp(const int* __restrict__ dst, const float* __restrict__ m,
                           float* __restrict__ sc, float* __restrict__ ssum, int E)
{
  int idx = blockIdx.x * blockDim.x + threadIdx.x;
  if (idx >= E * HEADS) return;
  int e = idx >> 2, h = idx & 3;
  int d = dst[e];
  float mm = m[(size_t)d * HEADS + h];
  if (!(mm > -1e29f)) mm = 0.0f;  // mirrors isfinite guard
  float w = __expf(sc[idx] - mm);
  sc[idx] = w;
  atomicAdd(&ssum[(size_t)d * HEADS + h], w);
}

__global__ void k_edge_agg(const int* __restrict__ src, const int* __restrict__ dst,
                           const float* __restrict__ sc, const float* __restrict__ ssum,
                           const float* __restrict__ V, float* __restrict__ agg, int E)
{
  int idx = blockIdx.x * blockDim.x + threadIdx.x;
  if (idx >= E * HID) return;
  int e = idx >> 6, c = idx & 63, h = c >> 4;
  int s = src[e], d = dst[e];
  float alpha = sc[(size_t)e * HEADS + h] / (ssum[(size_t)d * HEADS + h] + 1e-16f);
  atomicAdd(&agg[(size_t)d * HID + c], alpha * V[(size_t)s * HID + c]);
}

__global__ void k_relu(float* __restrict__ p, int n) {
  int idx = blockIdx.x * blockDim.x + threadIdx.x;
  if (idx < n) p[idx] = fmaxf(p[idx], 0.0f);
}

// ---------------------------------------------------------------------------
// Global mean pool + FC head.
// ---------------------------------------------------------------------------
__global__ void k_pool(const float* __restrict__ h, const int* __restrict__ batch,
                       float* __restrict__ pool, float* __restrict__ cnt, int N)
{
  int idx = blockIdx.x * blockDim.x + threadIdx.x;
  if (idx >= N * HID) return;
  int n = idx >> 6, c = idx & 63;
  int g = batch[n];
  atomicAdd(&pool[(size_t)g * HID + c], h[idx]);
  if (c == 0) atomicAdd(&cnt[g], 1.0f);
}

__global__ void k_fc(const float* __restrict__ pool, const float* __restrict__ cnt,
                     const float* __restrict__ Wfc, const float* __restrict__ bfc,
                     float* __restrict__ out, int B)
{
  int g = blockIdx.x * blockDim.x + threadIdx.x;
  if (g >= B) return;
  float c = fmaxf(cnt[g], 1.0f);
  float acc = bfc[0];
  const float* pr = pool + (size_t)g * HID;
#pragma unroll 8
  for (int i = 0; i < HID; i++) acc += (pr[i] / c) * Wfc[i];
  out[g] = acc;
}

// ---------------------------------------------------------------------------
extern "C" void kernel_launch(void* const* d_in, const int* in_sizes, int n_in,
                              void* d_out, int out_size, void* d_ws, size_t ws_size,
                              hipStream_t stream)
{
  (void)n_in; (void)ws_size;
  const float* x     = (const float*)d_in[0];
  const int*   src   = (const int*)d_in[1];
  const int*   dst   = (const int*)d_in[2];
  const int*   batch = (const int*)d_in[3];
  const int N   = in_sizes[0] / 15;
  const int E   = in_sizes[1];
  const int B   = out_size;   // num_graphs (device scalar unreadable under capture)
  const int FIN = 15;

  const float* W1[4]  = {(const float*)d_in[5],  (const float*)d_in[7],
                         (const float*)d_in[9],  (const float*)d_in[11]};
  const float* B1w[4] = {(const float*)d_in[6],  (const float*)d_in[8],
                         (const float*)d_in[10], (const float*)d_in[12]};
  const float* W2[4]  = {(const float*)d_in[13], (const float*)d_in[15],
                         (const float*)d_in[17], (const float*)d_in[19]};
  const float* B2w[4] = {(const float*)d_in[14], (const float*)d_in[16],
                         (const float*)d_in[18], (const float*)d_in[20]};
  const float* Wfc = (const float*)d_in[21];
  const float* bfc = (const float*)d_in[22];
  float* out = (float*)d_out;

  const int ntiles = (N + 15) / 16;

  // ---- workspace carve (256B aligned) ----
  char* p = (char*)d_ws;
  auto take = [&](size_t bytes) { char* r = p; p += (bytes + 255) & ~(size_t)255; return r; };
  _Float16* wht1[4]; for (int i = 0; i < 4; i++) wht1[i] = (_Float16*)take((size_t)32 * HID * 2);
  _Float16* wht2[4]; for (int i = 0; i < 4; i++) wht2[i] = (_Float16*)take((size_t)64 * HID * 2);
  _Float16* XH  = (_Float16*)take((size_t)ntiles * 16 * HID * 2); // f16 activations, tile-padded
  float* Qb   = (float*)take((size_t)N * HID * 4);
  float* Kb   = (float*)take((size_t)N * HID * 4);
  float* Vb   = (float*)take((size_t)N * HID * 4);
  float* AGG  = (float*)take((size_t)N * HID * 4);
  float* Mb   = (float*)take((size_t)N * HEADS * 4);
  float* Sb   = (float*)take((size_t)N * HEADS * 4);
  float* SC   = (float*)take((size_t)E * HEADS * 4);
  float* POOL = (float*)take((size_t)B * HID * 4);
  float* CNT  = (float*)take((size_t)B * 4);

  // ---- weights -> f16, transposed, padded ----
  for (int i = 0; i < 4; i++) {
    k_cvt_weight<<<(32 * HID + 255) / 256, 256, 0, stream>>>(W1[i], wht1[i], FIN, 32);
    k_cvt_weight<<<(64 * HID + 255) / 256, 256, 0, stream>>>(W2[i], wht2[i], HID, 64);
  }

  const int nh = N * HEADS, eh = E * HEADS, ec = E * HID, nc = N * HID;
  const int pgrid = ntiles < 3072 ? ntiles : 3072;   // persistent grid

  // ===== layer 1 =====
  k_cvt_x<<<((N * 32) + 255) / 256, 256, 0, stream>>>(x, XH, N);  // [N][32] f16
  k_linear4_wmma<32><<<pgrid, 128, 0, stream>>>(XH,
      wht1[0], wht1[1], wht1[2], wht1[3],
      B1w[0], B1w[1], B1w[2], B1w[3],
      Qb, Kb, Vb, AGG, N, ntiles);                    // AGG = skip1

  k_fill<<<(nh + 255) / 256, 256, 0, stream>>>(Mb, -1e30f, nh);
  k_fill<<<(nh + 255) / 256, 256, 0, stream>>>(Sb, 0.0f, nh);
  k_edge_score<<<(eh + 255) / 256, 256, 0, stream>>>(Qb, Kb, src, dst, SC, Mb, E);
  k_edge_exp  <<<(eh + 255) / 256, 256, 0, stream>>>(dst, Mb, SC, Sb, E);
  k_edge_agg  <<<(ec + 255) / 256, 256, 0, stream>>>(src, dst, SC, Sb, Vb, AGG, E);
  k_relu_cvt  <<<(nc + 255) / 256, 256, 0, stream>>>(AGG, XH, nc); // XH = h1 f16 [N][64]

  // ===== layer 2 (reads f16 h1; AGG is now a pure output) =====
  k_linear4_wmma<64><<<pgrid, 128, 0, stream>>>(XH,
      wht2[0], wht2[1], wht2[2], wht2[3],
      B2w[0], B2w[1], B2w[2], B2w[3],
      Qb, Kb, Vb, AGG, N, ntiles);                    // AGG = skip2

  k_fill<<<(nh + 255) / 256, 256, 0, stream>>>(Mb, -1e30f, nh);
  k_fill<<<(nh + 255) / 256, 256, 0, stream>>>(Sb, 0.0f, nh);
  k_edge_score<<<(eh + 255) / 256, 256, 0, stream>>>(Qb, Kb, src, dst, SC, Mb, E);
  k_edge_exp  <<<(eh + 255) / 256, 256, 0, stream>>>(dst, Mb, SC, Sb, E);
  k_edge_agg  <<<(ec + 255) / 256, 256, 0, stream>>>(src, dst, SC, Sb, Vb, AGG, E);
  k_relu      <<<(nc + 255) / 256, 256, 0, stream>>>(AGG, nc);     // AGG = h2 f32

  // ===== mean pool + FC =====
  k_fill<<<((B * HID) + 255) / 256, 256, 0, stream>>>(POOL, 0.0f, B * HID);
  k_fill<<<(B + 255) / 256, 256, 0, stream>>>(CNT, 0.0f, B);
  k_pool<<<(nc + 255) / 256, 256, 0, stream>>>(AGG, batch, POOL, CNT, N);
  k_fc  <<<(B + 255) / 256, 256, 0, stream>>>(POOL, CNT, Wfc, bfc, out, B);
}
